// SympleAgent_90434831385338
// MI455X (gfx1250) — matrix operand order repaired and checked by reading, not compiled, not run on hardware
//
#include <hip/hip_runtime.h>

#define TDEPTH 16
#define Hdim 256
#define Gdim 256
#define Vdim 128
#define NOPS 64
#define NNODES ((1 << (TDEPTH + 1)) - 1)   // 131071

typedef __attribute__((ext_vector_type(16))) __bf16 v16bf;
typedef __attribute__((ext_vector_type(8)))  __bf16 v8bf;
typedef __attribute__((ext_vector_type(8)))  float  v8f;
typedef __attribute__((ext_vector_type(4)))  unsigned int u32x4;
typedef __attribute__((ext_vector_type(8)))  int i32x8;
typedef __attribute__((ext_vector_type(4)))  int i32x4;

// Tensor Data Mover path (probe-confirmed builtins; arity differs by toolchain)
#if __has_builtin(__builtin_amdgcn_tensor_load_to_lds) && \
    __has_builtin(__builtin_amdgcn_s_wait_tensorcnt)
#define USE_TDM 1
#endif

// ---------------------------------------------------------------------------
// Fragment helpers (layouts per CDNA5 ISA 7.12.2, 16-bit A 16x32 / B 32x16)
// ---------------------------------------------------------------------------

// A fragment: lane L in [0,16) holds row M=L, K = k0+{0..7} and k0+{16..23};
// lane L in [16,32) holds row M=L-16, K = k0+{8..15} and k0+{24..31}.
__device__ __forceinline__ v16bf load_a_frag(const __bf16* rowPtr, int k0, int lane) {
  const int kb = (lane & 16) ? 8 : 0;
  const __bf16* p = rowPtr + k0 + kb;
  v8bf lo = *(const v8bf*)(p);
  v8bf hi = *(const v8bf*)(p + 16);
  v16bf r;
#pragma unroll
  for (int i = 0; i < 8; ++i) { r[i] = lo[i]; r[i + 8] = hi[i]; }
  return r;
}

// B fragment from pre-packed weights: 32 lanes x 16 contiguous bf16 per
// (kt, nt) 32x16 tile.  One 32B load.
__device__ __forceinline__ v16bf load_b_frag(const __bf16* Bpk, int kt, int nt,
                                             int ntilesN, int lane) {
  return *(const v16bf*)(Bpk + (((size_t)(kt * ntilesN + nt) * 32 + lane) << 4));
}

__device__ __forceinline__ v8f wmma_bf16(v16bf a, v16bf b, v8f c) {
  return __builtin_amdgcn_wmma_f32_16x16x32_bf16(false, a, false, b,
                                                 (short)0, c, false, false);
}

__device__ __forceinline__ float sigmf(float x) { return 1.f / (1.f + expf(-x)); }

// ---------------------------------------------------------------------------
// Prologue: f32 -> bf16 convert, and weight packing into B-fragment layout
// ---------------------------------------------------------------------------
__global__ void cvt_f32_bf16_kernel(const float* __restrict__ s,
                                    __bf16* __restrict__ d, int n) {
  int i = blockIdx.x * blockDim.x + threadIdx.x;
  if (i < n) d[i] = (__bf16)s[i];
}

// W is row-major [K x N]; out[tileIdx][lane][e] with
// ncol = nt*16 + (lane&15), k = kt*32 + (lane>>4)*8 + (e&7) + (e>>3)*16
__global__ void pack_b_kernel(const float* __restrict__ W,
                              __bf16* __restrict__ out, int K, int N) {
  int idx = blockIdx.x * blockDim.x + threadIdx.x;
  if (idx >= K * N) return;
  int e = idx & 15;
  int lane = (idx >> 4) & 31;
  int t = idx >> 9;
  int ntilesN = N >> 4;
  int kt = t / ntilesN, nt = t - kt * ntilesN;
  int ncol = nt * 16 + (lane & 15);
  int k = kt * 32 + ((lane >> 4) * 8) + (e & 7) + ((e >> 3) * 16);
  out[idx] = (__bf16)W[(size_t)k * N + ncol];
}

// ---------------------------------------------------------------------------
// TreeLSTM level: 16 nodes per block, 8 waves x (16 rows x 128 cols) WMMA.
// Child cell states are DMA'd into LDS by the Tensor Data Mover while the
// matrix cores run, hiding the phase-2 latency.
// ---------------------------------------------------------------------------
__global__ void __launch_bounds__(256) tree_level_kernel(
    const __bf16* __restrict__ A_bf, __bf16* __restrict__ h_bf,
    const __bf16* __restrict__ cIn, __bf16* __restrict__ cOut,
    const __bf16* __restrict__ Wx_pk, const __bf16* __restrict__ Ua_pk,
    const __bf16* __restrict__ Ub_pk, const float* __restrict__ bx,
    int n, int leaf) {
  extern __shared__ float smem[];
  float* Gs = smem;               // [16][1024]  g (+pa+pb for i/o/u, +pa for f)
  float* Fs = smem + 16 * 1024;   // [16][256]   g+pb for f chunk
#ifdef USE_TDM
  __bf16* cLds = (__bf16*)(smem + 16 * 1024 + 16 * 256);  // 32 child c rows (16KB)
#endif

  const int tid = threadIdx.x;
  const int lane = tid & 31;
  const int w = tid >> 5;
  const int tileBase = blockIdx.x * 16;
  const bool fchunk = (w >= 6);   // cols 768..1023

#ifdef USE_TDM
  // Kick off TDM: 32 contiguous child-c rows (2*tileBase .. 2*tileBase+31),
  // 16KB linear, global -> LDS.  1-D descriptor, data_size = 8B, 2048 elems.
  if (!leaf && w == 0) {
    unsigned long long ga =
        (unsigned long long)(uintptr_t)(cIn + (size_t)(2 * tileBase) * Hdim);
    unsigned la = (unsigned)(uintptr_t)cLds;      // low 32 bits = LDS offset
    u32x4 g0;
    g0[0] = 1u;                                   // count = 1 (valid descriptor)
    g0[1] = la;                                   // lds_addr
    g0[2] = (unsigned)ga;                         // global_addr[31:0]
    g0[3] = ((unsigned)(ga >> 32) & 0x01FFFFFFu)  // global_addr[56:32]
            | 0x80000000u;                        // type = 2 ("image")
    i32x8 g1;
    g1[0] = 0x00030000;      // workgroup_mask=0, data_size=3 (8 bytes)
    g1[1] = 0x08000000;      // tensor_dim0[15:0]=2048 in bits [31:16]
    g1[2] = 0x00010000;      // tensor_dim0[31:16]=0, tensor_dim1[15:0]=1
    g1[3] = 0x08000000;      // tensor_dim1[31:16]=0, tile_dim0=2048
    g1[4] = 0;               // tile_dim1=0, tile_dim2=0 (unused -> 1-D)
    g1[5] = 2048;            // tensor_dim0_stride[31:0]
    g1[6] = 0;               // stride[47:32]=0, tensor_dim1_stride[15:0]=0
    g1[7] = 0;
    i32x4 g2 = {0, 0, 0, 0};
    i32x4 g3 = {0, 0, 0, 0};
#if __clang_major__ >= 23
    i32x8 g4 = {0, 0, 0, 0, 0, 0, 0, 0};
    __builtin_amdgcn_tensor_load_to_lds(g0, g1, g2, g3, g4, 0);
#else
    __builtin_amdgcn_tensor_load_to_lds(g0, g1, g2, g3, 0);
#endif
  }
#endif

  // Per-lane A row pointers (row M = lane&15 of this tile).
  int m = lane & 15;
  int j = tileBase + m;
  if (j >= n) j = n - 1;
  const __bf16* xsRow = A_bf + (size_t)(n - 1 + j) * Vdim;
  const __bf16* haRow = h_bf + (size_t)(2 * n - 1 + 2 * j) * Hdim;
  const __bf16* hbRow = haRow + Hdim;

  v8f accA[8] = {};
  v8f accB[8] = {};

  // g = xs @ Wx   (K = 128)
#pragma unroll
  for (int kt = 0; kt < Vdim / 32; ++kt) {
    v16bf a = load_a_frag(xsRow, kt * 32, lane);
#pragma unroll
    for (int t = 0; t < 8; ++t) {
      v16bf b = load_b_frag(Wx_pk, kt, w * 8 + t, 64, lane);
      accA[t] = wmma_bf16(a, b, accA[t]);
    }
  }

  if (!leaf) {
    if (fchunk) {
#pragma unroll
      for (int t = 0; t < 8; ++t) accB[t] = accA[t];   // fb starts from g
    }
    // pa = ha @ Ua  (K = 256) -> accA
#pragma unroll
    for (int kt = 0; kt < Hdim / 32; ++kt) {
      v16bf a = load_a_frag(haRow, kt * 32, lane);
#pragma unroll
      for (int t = 0; t < 8; ++t) {
        v16bf b = load_b_frag(Ua_pk, kt, w * 8 + t, 64, lane);
        accA[t] = wmma_bf16(a, b, accA[t]);
      }
    }
    // pb = hb @ Ub  -> accA (i/o/u) or accB (f chunk)
#pragma unroll
    for (int kt = 0; kt < Hdim / 32; ++kt) {
      v16bf a = load_a_frag(hbRow, kt * 32, lane);
#pragma unroll
      for (int t = 0; t < 8; ++t) {
        v16bf b = load_b_frag(Ub_pk, kt, w * 8 + t, 64, lane);
        if (fchunk) accB[t] = wmma_bf16(a, b, accB[t]);
        else        accA[t] = wmma_bf16(a, b, accA[t]);
      }
    }
  }

  // Spill accumulators to LDS (D layout: lane -> col, vgpr v -> row)
  {
    int nl = lane & 15;
    int mb = (lane >> 4) * 8;
#pragma unroll
    for (int t = 0; t < 8; ++t) {
      int col = w * 128 + t * 16 + nl;
#pragma unroll
      for (int v = 0; v < 8; ++v) Gs[(mb + v) * 1024 + col] = accA[t][v];
      if (fchunk) {
#pragma unroll
        for (int v = 0; v < 8; ++v) Fs[(mb + v) * 256 + (col - 768)] = accB[t][v];
      }
    }
  }
#ifdef USE_TDM
  if (w == 0) __builtin_amdgcn_s_wait_tensorcnt(0);  // TDM done before barrier
#endif
  __syncthreads();

  // Gate nonlinearities + cell update: 16 (node,col) elems per thread.
  {
    int mm = tid >> 4;
    int jl = tileBase + mm;
    int jb = (tid & 15) * 16;
    if (jl < n) {
#ifdef USE_TDM
      const __bf16* caRow = cLds + (size_t)(2 * (jl - tileBase)) * Hdim;
#else
      const __bf16* caRow = cIn + (size_t)(2 * jl) * Hdim;
#endif
      const __bf16* cbRow = caRow + Hdim;
      __bf16* co = cOut + (size_t)jl * Hdim;
      __bf16* ho = h_bf + (size_t)(n - 1 + jl) * Hdim;
#pragma unroll 4
      for (int q = 0; q < 16; ++q) {
        int col = jb + q;
        float gi  = Gs[mm * 1024 + col]            + bx[col];
        float go  = Gs[mm * 1024 + Hdim + col]     + bx[Hdim + col];
        float gu  = Gs[mm * 1024 + 2 * Hdim + col] + bx[2 * Hdim + col];
        float gfa = Gs[mm * 1024 + 3 * Hdim + col] + bx[3 * Hdim + col];
        float gfb = Fs[mm * 256 + col]             + bx[3 * Hdim + col];
        float iv = sigmf(gi), ov = sigmf(go), uv = tanhf(gu);
        float ca = 0.f, cb = 0.f;
        if (!leaf) { ca = (float)caRow[col]; cb = (float)cbRow[col]; }
        float c = iv * uv + sigmf(gfa) * ca + sigmf(gfb) * cb;
        float h = ov * tanhf(c);
        co[col] = (__bf16)c;
        ho[col] = (__bf16)h;
      }
    }
  }
}

// ---------------------------------------------------------------------------
// Root global-LSTM cell + three small FFN heads + their softmaxes (1 block)
// ---------------------------------------------------------------------------
__device__ __forceinline__ void ffn_head(int tid, const float* rh, const float* hg,
                                         float* hid, const float* W1, const float* b1,
                                         const float* W2, const float* b2,
                                         int nout, float* lout) {
  float a = b1[tid];
  for (int k = 0; k < 256; ++k) a += rh[k] * W1[(size_t)k * 256 + tid];
  for (int k = 0; k < 256; ++k) a += hg[k] * W1[(size_t)(256 + k) * 256 + tid];
  hid[tid] = fmaxf(a, 0.f);
  __syncthreads();
  if (tid < nout) {
    float v = b2[tid];
    for (int k = 0; k < 256; ++k) v += hid[k] * W2[(size_t)k * nout + tid];
    lout[tid] = v;
  }
  __syncthreads();
}

__device__ __forceinline__ void softmax_write(const float* l, int n, float* o) {
  float M = l[0];
  for (int i = 1; i < n; ++i) M = fmaxf(M, l[i]);
  float s = 0.f;
  for (int i = 0; i < n; ++i) s += expf(l[i] - M);
  float inv = 1.f / s;
  for (int i = 0; i < n; ++i) o[i] = expf(l[i] - M) * inv;
}

__global__ void __launch_bounds__(256) heads_kernel(
    const __bf16* __restrict__ h_bf, const float* __restrict__ Wih,
    const float* __restrict__ bih, const float* __restrict__ bhh,
    const float* hiW1, const float* hib1, const float* hiW2, const float* hib2,
    const float* inW1, const float* inb1, const float* inW2, const float* inb2,
    const float* acW1, const float* acb1, const float* acW2, const float* acb2,
    const float* __restrict__ vmask, float* __restrict__ out,
    float* __restrict__ hg_f, __bf16* __restrict__ hg_bf) {
  __shared__ float rh[256], hg[256], hid[256], lh[4], li[5], le[64];
  int tid = threadIdx.x;
  rh[tid] = (float)h_bf[tid];   // root hidden = node 0
  __syncthreads();

  // gg = root_h @ Wih + bih + bhh ; split -> gi, gf(unused), gc, go
  float gi = bih[tid] + bhh[tid];
  float gc = bih[512 + tid] + bhh[512 + tid];
  float go = bih[768 + tid] + bhh[768 + tid];
  for (int k = 0; k < 256; ++k) {
    float r = rh[k];
    const float* row = Wih + (size_t)k * 1024;
    gi += r * row[tid];
    gc += r * row[512 + tid];
    go += r * row[768 + tid];
  }
  float cg = sigmf(gi) * tanhf(gc);
  float hv = sigmf(go) * tanhf(cg);
  hg[tid] = hv;
  hg_f[tid] = hv;
  hg_bf[tid] = (__bf16)hv;
  __syncthreads();

  ffn_head(tid, rh, hg, hid, hiW1, hib1, hiW2, hib2, 4, lh);
  ffn_head(tid, rh, hg, hid, inW1, inb1, inW2, inb2, 5, li);
  ffn_head(tid, rh, hg, hid, acW1, acb1, acW2, acb2, NOPS, le);
  if (tid < NOPS) le[tid] += logf(vmask[tid]);
  __syncthreads();
  if (tid == 0) {
    softmax_write(lh, 4, out);
    softmax_write(li, 5, out + 4);
    softmax_write(le, NOPS, out + 9);
  }
}

// ---------------------------------------------------------------------------
// Telemetry FFN: [h_glob | h_node] (512) -> 256 relu -> 1, WMMA + fused reduce
// ---------------------------------------------------------------------------
__global__ void __launch_bounds__(256) tel_kernel(
    const __bf16* __restrict__ hg_bf, const __bf16* __restrict__ h_bf,
    const __bf16* __restrict__ te1_pk, const float* __restrict__ te_b1,
    const float* __restrict__ te_W2, const float* __restrict__ te_b2,
    float* __restrict__ l_tel, int nnodes) {
  __shared__ float rowsum[16];
  const int tid = threadIdx.x;
  const int lane = tid & 31;
  const int w = tid >> 5;
  const int nodeBase = blockIdx.x * 16;
  if (tid < 16) rowsum[tid] = 0.f;
  __syncthreads();

  int m = lane & 15;
  int node = nodeBase + m;
  if (node >= nnodes) node = nnodes - 1;
  const __bf16* hrow = h_bf + (size_t)node * Hdim;

  v8f acc[2] = {};
#pragma unroll
  for (int kt = 0; kt < 16; ++kt) {
    v16bf a = (kt < 8) ? load_a_frag(hg_bf, kt * 32, lane)          // broadcast h_glob
                       : load_a_frag(hrow, (kt - 8) * 32, lane);    // node hidden
#pragma unroll
    for (int t = 0; t < 2; ++t) {
      v16bf b = load_b_frag(te1_pk, kt, w * 2 + t, 16, lane);
      acc[t] = wmma_bf16(a, b, acc[t]);
    }
  }

  // relu + second-layer dot (256 -> 1), per-lane partial over 8 rows
  int mb = (lane >> 4) * 8;
  float part[8];
#pragma unroll
  for (int v = 0; v < 8; ++v) part[v] = 0.f;
#pragma unroll
  for (int t = 0; t < 2; ++t) {
    int col = (w * 2 + t) * 16 + (lane & 15);
    float wv = te_W2[col];
    float bv = te_b1[col];
#pragma unroll
    for (int v = 0; v < 8; ++v) part[v] += fmaxf(acc[t][v] + bv, 0.f) * wv;
  }
#pragma unroll
  for (int v = 0; v < 8; ++v) atomicAdd(&rowsum[mb + v], part[v]);
  __syncthreads();
  if (tid < 16) {
    int nd = nodeBase + tid;
    if (nd < nnodes) l_tel[nd] = rowsum[tid] + te_b2[0];
  }
}

// ---------------------------------------------------------------------------
// Global softmax over l_tel (ordered-int max, f32 sum, write)
// ---------------------------------------------------------------------------
__device__ __forceinline__ unsigned ord_enc(float v) {
  unsigned u = __float_as_uint(v);
  return (u & 0x80000000u) ? ~u : (u | 0x80000000u);
}
__device__ __forceinline__ float ord_dec(unsigned u) {
  return (u & 0x80000000u) ? __uint_as_float(u ^ 0x80000000u) : __uint_as_float(~u);
}

__global__ void red_init_kernel(unsigned* maxb, float* sum) {
  if (threadIdx.x == 0 && blockIdx.x == 0) { *maxb = 0u; *sum = 0.f; }
}
__global__ void tel_max_kernel(const float* __restrict__ l, int n, unsigned* maxb) {
  int i = blockIdx.x * blockDim.x + threadIdx.x;
  if (i < n) atomicMax(maxb, ord_enc(l[i]));
}
__global__ void tel_sum_kernel(const float* __restrict__ l, int n,
                               const unsigned* __restrict__ maxb, float* sum) {
  int i = blockIdx.x * blockDim.x + threadIdx.x;
  if (i < n) atomicAdd(sum, expf(l[i] - ord_dec(*maxb)));
}
__global__ void tel_write_kernel(const float* __restrict__ l, int n,
                                 const unsigned* __restrict__ maxb,
                                 const float* __restrict__ sum,
                                 float* __restrict__ out) {
  int i = blockIdx.x * blockDim.x + threadIdx.x;
  if (i < n) out[i] = expf(l[i] - ord_dec(*maxb)) * (1.f / *sum);
}

// ---------------------------------------------------------------------------
// Host launcher
// ---------------------------------------------------------------------------
extern "C" void kernel_launch(void* const* d_in, const int* in_sizes, int n_in,
                              void* d_out, int out_size, void* d_ws, size_t ws_size,
                              hipStream_t stream) {
  const float* arg_hot = (const float*)d_in[0];
  const float* vmask   = (const float*)d_in[1];
  const float* Wx      = (const float*)d_in[2];
  const float* bx      = (const float*)d_in[3];
  const float* Ua      = (const float*)d_in[4];
  const float* Ub      = (const float*)d_in[5];
  const float* Wih     = (const float*)d_in[6];
  // d_in[7] = Whh multiplies zeros -> unused
  const float* bih     = (const float*)d_in[8];
  const float* bhh     = (const float*)d_in[9];
  const float* hiW1 = (const float*)d_in[10]; const float* hib1 = (const float*)d_in[11];
  const float* hiW2 = (const float*)d_in[12]; const float* hib2 = (const float*)d_in[13];
  const float* inW1 = (const float*)d_in[14]; const float* inb1 = (const float*)d_in[15];
  const float* inW2 = (const float*)d_in[16]; const float* inb2 = (const float*)d_in[17];
  const float* acW1 = (const float*)d_in[18]; const float* acb1 = (const float*)d_in[19];
  const float* acW2 = (const float*)d_in[20]; const float* acb2 = (const float*)d_in[21];
  const float* teW1 = (const float*)d_in[22]; const float* teb1 = (const float*)d_in[23];
  const float* teW2 = (const float*)d_in[24]; const float* teb2 = (const float*)d_in[25];
  float* out = (float*)d_out;

  // Workspace carve (256B aligned slices)
  size_t off = 0;
  char* base = (char*)d_ws;
  auto take = [&](size_t bytes) -> char* {
    char* r = base + off;
    off += (bytes + 255) & ~(size_t)255;
    return r;
  };
  __bf16* A_bf   = (__bf16*)take((size_t)NNODES * Vdim * 2);
  __bf16* h_bf   = (__bf16*)take((size_t)NNODES * Hdim * 2);
  __bf16* c0     = (__bf16*)take((size_t)65536 * Hdim * 2);
  __bf16* c1     = (__bf16*)take((size_t)65536 * Hdim * 2);
  __bf16* Wx_pk  = (__bf16*)take((size_t)Vdim * 4 * Hdim * 2);
  __bf16* Ua_pk  = (__bf16*)take((size_t)Hdim * 4 * Hdim * 2);
  __bf16* Ub_pk  = (__bf16*)take((size_t)Hdim * 4 * Hdim * 2);
  __bf16* te1_pk = (__bf16*)take((size_t)(Gdim + Hdim) * Hdim * 2);
  __bf16* hg_bf  = (__bf16*)take((size_t)Gdim * 2);
  float*  hg_f   = (float*) take((size_t)Gdim * 4);
  float*  l_tel  = (float*) take((size_t)NNODES * 4);
  unsigned* maxb = (unsigned*)take(4);
  float*  sum    = (float*) take(4);
  (void)ws_size; (void)n_in; (void)in_sizes; (void)out_size;

  // 1) convert activations, pack weights
  {
    int n = NNODES * Vdim;
    cvt_f32_bf16_kernel<<<(n + 255) / 256, 256, 0, stream>>>(arg_hot, A_bf, n);
  }
  pack_b_kernel<<<(Vdim * 4 * Hdim + 255) / 256, 256, 0, stream>>>(Wx, Wx_pk, Vdim, 4 * Hdim);
  pack_b_kernel<<<(Hdim * 4 * Hdim + 255) / 256, 256, 0, stream>>>(Ua, Ua_pk, Hdim, 4 * Hdim);
  pack_b_kernel<<<(Hdim * 4 * Hdim + 255) / 256, 256, 0, stream>>>(Ub, Ub_pk, Hdim, 4 * Hdim);
  pack_b_kernel<<<((Gdim + Hdim) * Hdim + 255) / 256, 256, 0, stream>>>(teW1, te1_pk, Gdim + Hdim, Hdim);

  // 2) TreeLSTM levels, leaves -> root (sequential dependency)
  // LDS: 16x1024 (Gs) + 16x256 (Fs) f32 + 16KB TDM-staged child c  = 96KB
  const size_t treeLds = (size_t)(16 * 1024 + 16 * 256) * sizeof(float) + 16384;
  for (int l = TDEPTH; l >= 0; --l) {
    int n = 1 << l;
    __bf16* cO = (l & 1) ? c1 : c0;
    const __bf16* cI = (l & 1) ? c0 : c1;
    int blocks = (n + 15) / 16;
    tree_level_kernel<<<blocks, 256, treeLds, stream>>>(
        A_bf, h_bf, cI, cO, Wx_pk, Ua_pk, Ub_pk, bx, n, (l == TDEPTH) ? 1 : 0);
  }

  // 3) root global cell + small heads + their softmaxes
  heads_kernel<<<1, 256, 0, stream>>>(h_bf, Wih, bih, bhh,
                                      hiW1, hib1, hiW2, hib2,
                                      inW1, inb1, inW2, inb2,
                                      acW1, acb1, acW2, acb2,
                                      vmask, out, hg_f, hg_bf);

  // 4) telemetry FFN over all nodes (WMMA)
  tel_kernel<<<(NNODES + 15) / 16, 256, 0, stream>>>(
      hg_bf, h_bf, te1_pk, teb1, teW2, teb2, l_tel, NNODES);

  // 5) global softmax over l_tel -> out[73 ..]
  red_init_kernel<<<1, 32, 0, stream>>>(maxb, sum);
  tel_max_kernel<<<(NNODES + 255) / 256, 256, 0, stream>>>(l_tel, NNODES, maxb);
  tel_sum_kernel<<<(NNODES + 255) / 256, 256, 0, stream>>>(l_tel, NNODES, maxb, sum);
  tel_write_kernel<<<(NNODES + 255) / 256, 256, 0, stream>>>(l_tel, NNODES, maxb, sum,
                                                             out + 9 + NOPS);
}
